// CoarseSkeletonHead_18236431139318
// MI455X (gfx1250) — matrix operand order repaired and checked by reading, not compiled, not run on hardware
//
#include <hip/hip_runtime.h>

typedef __attribute__((ext_vector_type(16))) __bf16 v16bf;
typedef __attribute__((ext_vector_type(8)))  float  v8f;

constexpr int kE  = 512;
constexpr int kS  = 128;
constexpr int kJ  = 24;
constexpr int kH  = 8;
constexpr int kHD = 64;

// LDS layout (bytes)
constexpr int kVtOff = 0;          // Vt bf16 [512][128]            = 131072
constexpr int kScOff = 131072;     // scores f32 [8*24][128]        =  98304
                                   //   reuse: attn bf16 @ +49152 (48K), h f32 @ +0 (48K)
constexpr int kChOff = 229376;     // chunk region                  =  65536
                                   //   zch/o_bf @ +0 (32K), Kch/jf_bf @ +32768 (32K)
constexpr int kLds   = 294912;     // total (<= 320KB WGP LDS)

__device__ __forceinline__ v8f vzero8() {
  v8f z;
#pragma unroll
  for (int i = 0; i < 8; ++i) z[i] = 0.0f;
  return z;
}

__device__ __forceinline__ v8f wmma_bf16(v16bf a, v16bf b, v8f c) {
  // D = A(16x32 bf16) * B(32x16 bf16) + C(16x16 f32)
  return __builtin_amdgcn_wmma_f32_16x16x32_bf16(
      /*neg_a=*/false, a, /*neg_b=*/false, b,
      /*c_mod=*/(short)0, c, /*reuse_a=*/false, /*reuse_b=*/false);
}

// A-matrix fragment (16x32 bf16): element e -> k = (e&7) + 8*lh + 16*(e>>3)
__device__ __forceinline__ v16bf ldA_bf16(const __bf16* base, int stride,
                                          int row, int col0, int lh) {
  const __bf16* p = base + row * stride + col0 + 8 * lh;
  v16bf f;
#pragma unroll
  for (int e = 0; e < 8; ++e) f[e] = p[e];
#pragma unroll
  for (int e = 0; e < 8; ++e) f[8 + e] = p[16 + e];
  return f;
}

// Guarded A fragment: rows >= rowlim read as zero
__device__ __forceinline__ v16bf ldA_bf16_lim(const __bf16* base, int stride,
                                              int row, int col0, int lh, int rowlim) {
  v16bf f;
  if (row < rowlim) {
    const __bf16* p = base + row * stride + col0 + 8 * lh;
#pragma unroll
    for (int e = 0; e < 8; ++e) f[e] = p[e];
#pragma unroll
    for (int e = 0; e < 8; ++e) f[8 + e] = p[16 + e];
  } else {
#pragma unroll
    for (int e = 0; e < 16; ++e) f[e] = (__bf16)0.0f;
  }
  return f;
}

// B-matrix fragment (32x16 bf16): lane = column n; element e -> k = (e&7) + 8*lh + 16*(e>>3)
__device__ __forceinline__ v16bf ldB_bf16(const __bf16* base, int stride,
                                          int row, int col0, int lh) {
  const __bf16* p = base + row * stride + col0 + 8 * lh;
  v16bf f;
#pragma unroll
  for (int e = 0; e < 8; ++e) f[e] = p[e];
#pragma unroll
  for (int e = 0; e < 8; ++e) f[8 + e] = p[16 + e];
  return f;
}

// ---------------------------------------------------------------------------
// Prep kernel 1: q = (joint_queries @ wq^T + bq) / sqrt(64), bf16, padded to 32 rows
__global__ void csh_prep_q(const float* __restrict__ jq,
                           const float* __restrict__ in_w,
                           const float* __restrict__ in_b,
                           __bf16* __restrict__ q_ws) {
  int idx = blockIdx.x * 256 + threadIdx.x;     // 32*512
  if (idx >= 32 * kE) return;
  int j = idx >> 9, n = idx & (kE - 1);
  float acc = 0.0f;
  if (j < kJ) {
    const float* a = jq + j * kE;
    const float* w = in_w + n * kE;              // wq rows 0..511
    for (int e = 0; e < kE; ++e) acc += a[e] * w[e];
    acc = (acc + in_b[n]) * 0.125f;              // 1/sqrt(HD)
  }
  q_ws[idx] = (__bf16)acc;
}

// Prep kernel 2: convert wk, wv, out_proj_w, w1 to bf16 (row-major n x k)
__global__ void csh_prep_w(const float* __restrict__ in_w,
                           const float* __restrict__ ow,
                           const float* __restrict__ w1,
                           __bf16* __restrict__ dst) {
  int idx = blockIdx.x * 256 + threadIdx.x;      // 4 * 262144
  int m = idx >> 18;
  int off = idx & 262143;
  float v;
  if (m == 0)      v = in_w[512 * kE + off];     // wk
  else if (m == 1) v = in_w[1024 * kE + off];    // wv
  else if (m == 2) v = ow[off];                  // out_proj_w
  else             v = w1[off];                  // w1
  dst[idx] = (__bf16)v;
}

// ---------------------------------------------------------------------------
// Fused per-batch kernel: KV proj -> attention -> out_proj -> MLP -> skeleton
__global__ __launch_bounds__(256, 1)
void csh_fused(const float* __restrict__ z,
               const float* __restrict__ in_b,
               const float* __restrict__ out_b,
               const float* __restrict__ b1,
               const float* __restrict__ lng,
               const float* __restrict__ lnb,
               const float* __restrict__ w2,
               const float* __restrict__ b2,
               const int*   __restrict__ parent,
               const __bf16* __restrict__ q_ws,
               const __bf16* __restrict__ wk_ws,
               const __bf16* __restrict__ wv_ws,
               const __bf16* __restrict__ wo_ws,
               const __bf16* __restrict__ w1_ws,
               float* __restrict__ out) {
  extern __shared__ char smem[];
  __bf16* Vt    = (__bf16*)(smem + kVtOff);             // [512][128] bf16
  float*  sc    = (float*)(smem + kScOff);              // [8*24][128] f32
  __bf16* abuf  = (__bf16*)(smem + kScOff + 49152);     // attn bf16 [8*24][128]
  float*  hb    = (float*)(smem + kScOff);              // [24][512] f32 (reuse)
  __bf16* zch   = (__bf16*)(smem + kChOff);             // [32][512] bf16
  __bf16* Kch   = (__bf16*)(smem + kChOff + 32768);     // [32][512] bf16
  __bf16* o_bf  = (__bf16*)(smem + kChOff);             // [32][512] bf16 (reuse)
  __bf16* jf_bf = (__bf16*)(smem + kChOff + 32768);     // [32][512] bf16 (reuse)
  float*  raw   = (float*)(smem + kChOff);              // [24][4] f32 (reuse)
  float*  offs  = (float*)(smem + kChOff + 512);        // [24][3] f32 (reuse)

  const int B    = gridDim.x;
  const int b    = blockIdx.x;
  const int tid  = threadIdx.x;
  const int wave = tid >> 5;
  const int lane = tid & 31;
  const int lh   = lane >> 4;
  const int lr   = lane & 15;
  const float* bk = in_b + kE;
  const float* bv = in_b + 2 * kE;

  // ===== Phase A: K/V projection + scores, per 32-row s-chunk =====
  for (int scnk = 0; scnk < 4; ++scnk) {
    const float* zsrc = z + ((size_t)b * kS + scnk * 32) * kE;
    // stage z chunk as bf16
    for (int i = tid; i < (32 * kE) / 4; i += 256) {
      float4 f = ((const float4*)zsrc)[i];
      __bf16* d = zch + i * 4;
      d[0] = (__bf16)f.x; d[1] = (__bf16)f.y;
      d[2] = (__bf16)f.z; d[3] = (__bf16)f.w;
    }
    __syncthreads();

    // K chunk: (32 x 512) = z_chunk @ wk^T + bk  -> Kch bf16
    // one B fragment feeds both M-tiles
    for (int nt = wave; nt < 32; nt += 8) {
      int n0 = nt * 16;
      v8f acc0 = vzero8(), acc1 = vzero8();
      for (int kk = 0; kk < kE; kk += 32) {
        v16bf bb = ldB_bf16(wk_ws, kE, n0 + lr, kk, lh);
        v16bf a0 = ldA_bf16(zch, kE, lr, kk, lh);
        v16bf a1 = ldA_bf16(zch, kE, 16 + lr, kk, lh);
        acc0 = wmma_bf16(a0, bb, acc0);
        acc1 = wmma_bf16(a1, bb, acc1);
      }
      float bias = bk[n0 + lr];
#pragma unroll
      for (int v = 0; v < 8; ++v) {
        Kch[(v + 8 * lh) * kE + n0 + lr]      = (__bf16)(acc0[v] + bias);
        Kch[(16 + v + 8 * lh) * kE + n0 + lr] = (__bf16)(acc1[v] + bias);
      }
    }
    __syncthreads();

    // scores: per head, q(24x64, padded 32) @ K_chunk^T -> sc f32
    for (int g = wave; g < 16; g += 8) {
      int h = g >> 1, st = g & 1;
      int n0 = st * 16;
      v8f acc0 = vzero8(), acc1 = vzero8();
      for (int kd = 0; kd < kHD; kd += 32) {
        v16bf bb = ldB_bf16(Kch, kE, n0 + lr, h * kHD + kd, lh);
        v16bf a0 = ldA_bf16(q_ws, kE, lr, h * kHD + kd, lh);
        v16bf a1 = ldA_bf16(q_ws, kE, 16 + lr, h * kHD + kd, lh);
        acc0 = wmma_bf16(a0, bb, acc0);
        acc1 = wmma_bf16(a1, bb, acc1);
      }
      int sg = scnk * 32 + n0 + lr;
#pragma unroll
      for (int v = 0; v < 8; ++v) {
        int j0 = v + 8 * lh;
        sc[(h * kJ + j0) * kS + sg] = acc0[v];
        int j1 = 16 + v + 8 * lh;
        if (j1 < kJ) sc[(h * kJ + j1) * kS + sg] = acc1[v];
      }
    }

    // V chunk: (32 x 512) = z_chunk @ wv^T + bv -> Vt (transposed) bf16
    for (int nt = wave; nt < 32; nt += 8) {
      int n0 = nt * 16;
      v8f acc0 = vzero8(), acc1 = vzero8();
      for (int kk = 0; kk < kE; kk += 32) {
        v16bf bb = ldB_bf16(wv_ws, kE, n0 + lr, kk, lh);
        v16bf a0 = ldA_bf16(zch, kE, lr, kk, lh);
        v16bf a1 = ldA_bf16(zch, kE, 16 + lr, kk, lh);
        acc0 = wmma_bf16(a0, bb, acc0);
        acc1 = wmma_bf16(a1, bb, acc1);
      }
      float bias = bv[n0 + lr];
      int col = n0 + lr;
      int sbase = scnk * 32;
#pragma unroll
      for (int v = 0; v < 8; ++v) {
        Vt[col * kS + sbase + v + 8 * lh]      = (__bf16)(acc0[v] + bias);
        Vt[col * kS + sbase + 16 + v + 8 * lh] = (__bf16)(acc1[v] + bias);
      }
    }
    __syncthreads();
  }

  // ===== Phase B: softmax over s (192 rows), write attn as bf16 =====
  if (tid < kH * kJ) {
    float* row = sc + tid * kS;
    __bf16* arow = abuf + tid * kS;
    float mx = -1e30f;
    for (int i = 0; i < kS; ++i) mx = fmaxf(mx, row[i]);
    float sum = 0.0f;
    float e[kS / 32];  // unused placeholder removed below
    (void)e;
    float tmp;
    for (int i = 0; i < kS; ++i) { tmp = __expf(row[i] - mx); row[i] = tmp; sum += tmp; }
    float inv = 1.0f / sum;
    for (int i = 0; i < kS; ++i) arow[i] = (__bf16)(row[i] * inv);
  }
  __syncthreads();

  // ===== Phase C: o = attn @ V (per head) -> o_bf bf16 [32][512] =====
  for (int g = wave; g < 32; g += 8) {
    int h = g & 7, dt = g >> 3;
    int ncol = h * kHD + dt * 16;
    const __bf16* ab = abuf + h * kJ * kS;
    v8f acc0 = vzero8(), acc1 = vzero8();
    for (int ss = 0; ss < kS; ss += 32) {
      v16bf bb = ldB_bf16(Vt, kS, ncol + lr, ss, lh);
      v16bf a0 = ldA_bf16_lim(ab, kS, lr, ss, lh, kJ);
      v16bf a1 = ldA_bf16_lim(ab, kS, 16 + lr, ss, lh, kJ);
      acc0 = wmma_bf16(a0, bb, acc0);
      acc1 = wmma_bf16(a1, bb, acc1);
    }
#pragma unroll
    for (int v = 0; v < 8; ++v) {
      o_bf[(v + 8 * lh) * kE + ncol + lr]      = (__bf16)acc0[v];
      o_bf[(16 + v + 8 * lh) * kE + ncol + lr] = (__bf16)acc1[v];  // rows >=24 are zero
    }
  }
  __syncthreads();

  // ===== Phase D: joint_features = o @ out_proj_w^T + b -> jf_bf (zero-padded) =====
  for (int nt = wave; nt < 32; nt += 8) {
    int n0 = nt * 16;
    v8f acc0 = vzero8(), acc1 = vzero8();
    for (int kk = 0; kk < kE; kk += 32) {
      v16bf bb = ldB_bf16(wo_ws, kE, n0 + lr, kk, lh);
      v16bf a0 = ldA_bf16(o_bf, kE, lr, kk, lh);
      v16bf a1 = ldA_bf16(o_bf, kE, 16 + lr, kk, lh);
      acc0 = wmma_bf16(a0, bb, acc0);
      acc1 = wmma_bf16(a1, bb, acc1);
    }
    float bias = out_b[n0 + lr];
#pragma unroll
    for (int v = 0; v < 8; ++v) {
      jf_bf[(v + 8 * lh) * kE + n0 + lr] = (__bf16)(acc0[v] + bias);
      int j1 = 16 + v + 8 * lh;
      jf_bf[j1 * kE + n0 + lr] = (__bf16)((j1 < kJ) ? (acc1[v] + bias) : 0.0f);
    }
  }
  __syncthreads();

  // ===== Phase E: h = jf @ w1^T + b1 -> hb f32 (24 rows) =====
  for (int nt = wave; nt < 32; nt += 8) {
    int n0 = nt * 16;
    v8f acc0 = vzero8(), acc1 = vzero8();
    for (int kk = 0; kk < kE; kk += 32) {
      v16bf bb = ldB_bf16(w1_ws, kE, n0 + lr, kk, lh);
      v16bf a0 = ldA_bf16(jf_bf, kE, lr, kk, lh);
      v16bf a1 = ldA_bf16(jf_bf, kE, 16 + lr, kk, lh);
      acc0 = wmma_bf16(a0, bb, acc0);
      acc1 = wmma_bf16(a1, bb, acc1);
    }
    float bias = b1[n0 + lr];
#pragma unroll
    for (int v = 0; v < 8; ++v) {
      hb[(v + 8 * lh) * kE + n0 + lr] = acc0[v] + bias;
      int j1 = 16 + v + 8 * lh;
      if (j1 < kJ) hb[j1 * kE + n0 + lr] = acc1[v] + bias;
    }
  }
  __syncthreads();

  // ===== Phase F: LayerNorm + SiLU (one wave per row) =====
  for (int j = wave; j < kJ; j += 8) {
    float* hr = hb + j * kE;
    float s = 0.0f, s2 = 0.0f;
    float vals[16];
#pragma unroll
    for (int i = 0; i < 16; ++i) {
      float x = hr[lane + 32 * i];
      vals[i] = x; s += x; s2 += x * x;
    }
#pragma unroll
    for (int m = 16; m >= 1; m >>= 1) {
      s  += __shfl_xor(s, m, 32);
      s2 += __shfl_xor(s2, m, 32);
    }
    float mu   = s * (1.0f / 512.0f);
    float var  = s2 * (1.0f / 512.0f) - mu * mu;
    float rstd = rsqrtf(var + 1e-5f);
#pragma unroll
    for (int i = 0; i < 16; ++i) {
      int e = lane + 32 * i;
      float x = (vals[i] - mu) * rstd * lng[e] + lnb[e];
      x = x / (1.0f + __expf(-x));
      hr[e] = x;
    }
  }
  __syncthreads();

  // ===== Phase G: raw = h @ w2^T + b2  (24x4) =====
  if (tid < kJ * 4) {
    int j = tid >> 2, c = tid & 3;
    const float* wr = w2 + c * kE;
    const float* hr = hb + j * kE;
    float acc = b2[c];
    for (int e = 0; e < kE; ++e) acc += hr[e] * wr[e];
    raw[j * 4 + c] = acc;
  }
  __syncthreads();

  // ===== Phase H: direction / softplus length / offsets / joints =====
  const size_t offBase = (size_t)B * kJ * 3;
  const size_t lenBase = (size_t)B * kJ * 6;
  if (tid < kJ) {
    int j = tid;
    float ox = 0.0f, oy = 0.0f, oz = 0.0f;
    if (j > 0) {
      float dx = raw[j * 4 + 0], dy = raw[j * 4 + 1], dz = raw[j * 4 + 2];
      float lraw = raw[j * 4 + 3];
      float nrm = sqrtf(dx * dx + dy * dy + dz * dz);
      float inv = 1.0f / fmaxf(nrm, 1e-6f);
      float len = fmaxf(lraw, 0.0f) + log1pf(__expf(-fabsf(lraw)));
      ox = dx * inv * len; oy = dy * inv * len; oz = dz * inv * len;
      out[lenBase + (size_t)b * (kJ - 1) + (j - 1)] = len;
    }
    offs[j * 3 + 0] = ox; offs[j * 3 + 1] = oy; offs[j * 3 + 2] = oz;
    size_t ob = offBase + ((size_t)b * kJ + j) * 3;
    out[ob + 0] = ox; out[ob + 1] = oy; out[ob + 2] = oz;
  }
  __syncthreads();
  if (tid < kJ) {
    int j = tid;
    float jx = 0.0f, jy = 0.0f, jz = 0.0f;
    int cur = j;
    while (cur >= 0) {
      jx += offs[cur * 3 + 0];
      jy += offs[cur * 3 + 1];
      jz += offs[cur * 3 + 2];
      cur = parent[cur];
    }
    size_t ob = ((size_t)b * kJ + j) * 3;
    out[ob + 0] = jx; out[ob + 1] = jy; out[ob + 2] = jz;
  }
}

// ---------------------------------------------------------------------------
extern "C" void kernel_launch(void* const* d_in, const int* in_sizes, int n_in,
                              void* d_out, int out_size, void* d_ws, size_t ws_size,
                              hipStream_t stream) {
  const float* z    = (const float*)d_in[0];
  const float* jq   = (const float*)d_in[1];
  const float* in_w = (const float*)d_in[2];
  const float* in_b = (const float*)d_in[3];
  const float* ow   = (const float*)d_in[4];
  const float* ob   = (const float*)d_in[5];
  const float* w1   = (const float*)d_in[6];
  const float* b1   = (const float*)d_in[7];
  const float* lng  = (const float*)d_in[8];
  const float* lnb  = (const float*)d_in[9];
  const float* w2   = (const float*)d_in[10];
  const float* b2   = (const float*)d_in[11];
  const int*   par  = (const int*)d_in[12];
  float* out = (float*)d_out;

  __bf16* q_ws  = (__bf16*)d_ws;                 // 32*512
  __bf16* wk_ws = q_ws + 32 * kE;                // 512*512
  __bf16* wv_ws = wk_ws + kE * kE;
  __bf16* wo_ws = wv_ws + kE * kE;
  __bf16* w1_ws = wo_ws + kE * kE;

  int B = in_sizes[0] / (kS * kE);               // 1024

  csh_prep_q<<<(32 * kE) / 256, 256, 0, stream>>>(jq, in_w, in_b, q_ws);
  csh_prep_w<<<(4 * kE * kE) / 256, 256, 0, stream>>>(in_w, ow, w1, wk_ws);

  (void)hipFuncSetAttribute((const void*)csh_fused,
                            hipFuncAttributeMaxDynamicSharedMemorySize, kLds);
  csh_fused<<<B, 256, kLds, stream>>>(z, in_b, ob, b1, lng, lnb, w2, b2, par,
                                      q_ws, wk_ws, wv_ws, wo_ws, w1_ws, out);
}
